// DeformAttn_28243704938552
// MI455X (gfx1250) — matrix operand order repaired and checked by reading, not compiled, not run on hardware
//
#include <hip/hip_runtime.h>
#include <hip/hip_bf16.h>
#include <math.h>

// ---------------------------------------------------------------------------
// Problem constants (from reference)
// ---------------------------------------------------------------------------
#define BB   2
#define CC   192
#define HH   96
#define WW   96
#define HWP  (HH * WW)          // 9216 pixels per image
#define NPIX (BB * HWP)         // 18432 pixels total
#define HEADS 12
#define DG    12
#define CG    16                // channels per group == head dim
#define KK    9
#define KH    3
#define KW    3
#define PADR  1
#define MLP_H 384

typedef __attribute__((ext_vector_type(16))) _Float16 v16h;
typedef __attribute__((ext_vector_type(8)))  _Float16 v8h;
typedef __attribute__((ext_vector_type(8)))  float    v8f;

// ---------------------------------------------------------------------------
// Fused GEMM:  Y = act( X^T W + bias ) (+ residual)
// X layout: [B, Cin, HW] (channel-major).  W: [Cin, Cout] row-major.
// Tile: M = 128 pixels (8 waves x 16 rows), N = 192 per grid.y, K-step 32.
// LDS double-buffered; next tile's global loads are issued before the WMMA
// chain so v_wmma latency hides them.  One barrier per K-step.
// olayout: 0 -> Y[B, Cout, HW] (channel-major)
//          1 -> Y[((b*DG + g)*HW + hw)*16 + cg]   (group-pixel-major, Cout=192)
// ---------------------------------------------------------------------------
__global__ __launch_bounds__(256)
void gemm_nchw_wmma(const float* __restrict__ X, int Cin,
                    const float* __restrict__ Wt, const float* __restrict__ bias,
                    int Cout, float* __restrict__ Y,
                    const float* __restrict__ residual, int gelu, int olayout)
{
    const int tilesPerImg = HWP / 128;              // 72 (exact)
    const int tile = blockIdx.x;
    const int b    = tile / tilesPerImg;
    const int hw0  = (tile % tilesPerImg) * 128;
    const int nblk = blockIdx.y * 192;

    const int tid  = threadIdx.x;
    const int lane = tid & 31;
    const int wv   = tid >> 5;                      // 8 waves
    const int mloc = wv * 16;

    __shared__ _Float16 ldsA[2][128 * 32];
    __shared__ _Float16 ldsB[2][32 * 192];

    // ---- per-thread quad maps (computed once; int offsets, no 64-bit mul) ----
    int opA[4], lsA[4];                             // A: 4 quads of 4 pixels
    #pragma unroll
    for (int i = 0; i < 4; ++i) {
        int q  = tid + i * 256;                     // 1024 quads
        int c  = q >> 5;                            // channel 0..31
        int p4 = (q & 31) << 2;                     // pixel 0..124 step 4
        opA[i] = c * HWP + p4;
        lsA[i] = p4 * 32 + c;                       // +32 halfs per pixel
    }
    int opB[6], lsB[6];                             // B: 6 quads of 4 outputs
    #pragma unroll
    for (int i = 0; i < 6; ++i) {
        int q  = tid + i * 256;                     // 1536 quads
        int kk = q / 48;                            // K row 0..31
        int dq = (q - kk * 48) << 2;                // d 0..188 step 4
        opB[i] = kk * Cout + nblk + dq;
        int nt = dq >> 4;
        int nn = dq & 15;
        int ln = nn + ((kk >> 4) << 4);
        int hh = kk & 15;
        lsB[i] = (nt * 32 + ln) * 16 + hh;          // +16 halfs per successive d
    }

    const float* Xg = X + (size_t)b * Cin * HWP + hw0;

    float ra[16], rb[24];
    // ---- prologue: fetch tile 0 (batched float4 loads), stage into buf 0 ----
    #pragma unroll
    for (int i = 0; i < 4; ++i) {
        float4 t = *(const float4*)(Xg + opA[i]);
        ra[4*i+0] = t.x; ra[4*i+1] = t.y; ra[4*i+2] = t.z; ra[4*i+3] = t.w;
    }
    #pragma unroll
    for (int i = 0; i < 6; ++i) {
        float4 t = *(const float4*)(Wt + opB[i]);
        rb[4*i+0] = t.x; rb[4*i+1] = t.y; rb[4*i+2] = t.z; rb[4*i+3] = t.w;
    }
    #pragma unroll
    for (int i = 0; i < 4; ++i)
        #pragma unroll
        for (int j = 0; j < 4; ++j)
            ldsA[0][lsA[i] + 32 * j] = (_Float16)ra[4*i+j];
    #pragma unroll
    for (int i = 0; i < 6; ++i)
        #pragma unroll
        for (int j = 0; j < 4; ++j)
            ldsB[0][lsB[i] + 16 * j] = (_Float16)rb[4*i+j];

    v8f acc[12] = {};
    const int nsteps = Cin / 32;
    int cur = 0;

    for (int s = 0; s < nsteps; ++s) {
        __syncthreads();                            // buf[cur] ready for all waves

        const bool hasNext = (s + 1) < nsteps;
        if (hasNext) {
            // issue next tile's global loads now; WMMA below hides latency
            const float* Xk = Xg + (s + 1) * 32 * HWP;
            const float* Wk = Wt + (s + 1) * 32 * Cout;
            #pragma unroll
            for (int i = 0; i < 4; ++i) {
                float4 t = *(const float4*)(Xk + opA[i]);
                ra[4*i+0] = t.x; ra[4*i+1] = t.y; ra[4*i+2] = t.z; ra[4*i+3] = t.w;
            }
            #pragma unroll
            for (int i = 0; i < 6; ++i) {
                float4 t = *(const float4*)(Wk + opB[i]);
                rb[4*i+0] = t.x; rb[4*i+1] = t.y; rb[4*i+2] = t.z; rb[4*i+3] = t.w;
            }
        }

        // ---- fragments from buf[cur] ----
        // A (16-bit 16x32): lanes 0-15 K{0-15}, lanes 16-31 K{16-31}
        int m  = lane & 15;
        int hi = lane >> 4;
        const _Float16* pa = &ldsA[cur][(mloc + m) * 32 + hi * 16];
        v8h a0 = *(const v8h*)(pa);
        v8h a1 = *(const v8h*)(pa + 8);
        v16h afrag = __builtin_shufflevector(a0, a1,
                        0, 1, 2, 3, 4, 5, 6, 7, 8, 9, 10, 11, 12, 13, 14, 15);

        #pragma unroll
        for (int nt = 0; nt < 12; ++nt) {
            v16h bfrag = *(const v16h*)&ldsB[cur][(nt * 32 + lane) * 16];
            acc[nt] = __builtin_amdgcn_wmma_f32_16x16x32_f16(
                          false, afrag, false, bfrag,
                          (short)0, acc[nt], false, false);
        }

        if (hasNext) {                              // stage next tile into buf^1
            int nxt = cur ^ 1;
            #pragma unroll
            for (int i = 0; i < 4; ++i)
                #pragma unroll
                for (int j = 0; j < 4; ++j)
                    ldsA[nxt][lsA[i] + 32 * j] = (_Float16)ra[4*i+j];
            #pragma unroll
            for (int i = 0; i < 6; ++i)
                #pragma unroll
                for (int j = 0; j < 4; ++j)
                    ldsB[nxt][lsB[i] + 16 * j] = (_Float16)rb[4*i+j];
        }
        cur ^= 1;
    }

    // ---- epilogue: v8f element i -> pixel M = i + 8*hi, channel N = lane&15 ----
    const int nn = lane & 15;
    const int hi = lane >> 4;
    #pragma unroll
    for (int nt = 0; nt < 12; ++nt) {
        int d = nblk + nt * 16 + nn;
        float bvv = bias[d];
        float vals[8];
        #pragma unroll
        for (int i = 0; i < 8; ++i) {
            float val = acc[nt][i] + bvv;
            if (gelu) val = 0.5f * val * (1.0f + erff(val * 0.70710678118654752f));
            vals[i] = val;
        }
        if (olayout == 0) {
            size_t obase = ((size_t)b * Cout + d) * HWP + hw0 + mloc + hi * 8;
            if (residual) {
                float4 r0 = *(const float4*)(residual + obase);
                float4 r1 = *(const float4*)(residual + obase + 4);
                vals[0] += r0.x; vals[1] += r0.y; vals[2] += r0.z; vals[3] += r0.w;
                vals[4] += r1.x; vals[5] += r1.y; vals[6] += r1.z; vals[7] += r1.w;
            }
            float4 o0 = make_float4(vals[0], vals[1], vals[2], vals[3]);
            float4 o1 = make_float4(vals[4], vals[5], vals[6], vals[7]);
            *(float4*)(Y + obase) = o0;
            *(float4*)(Y + obase + 4) = o1;
        } else {
            // [((b*DG + g)*HW + hw)*16 + cg] ; g = nt, cg = nn (nblk==0, Cout==192)
            size_t obase = (((size_t)b * DG + nt) * HWP + hw0 + mloc + hi * 8) * CG + nn;
            #pragma unroll
            for (int i = 0; i < 8; ++i)
                Y[obase + (size_t)i * CG] = vals[i];
        }
    }
}

// ---------------------------------------------------------------------------
// Deformable sampling + softmax attention.
// q/k/v projections are in [B, G, HW, 16] layout -> each bilinear corner is
// four float4 loads (64B contiguous) instead of 16 scalar gathers.
// One thread per (pixel, head); head partition == group partition.
// ---------------------------------------------------------------------------
__global__ __launch_bounds__(256)
void deform_attn_kernel(const float* __restrict__ qp, const float* __restrict__ kp,
                        const float* __restrict__ vp, const float* __restrict__ offs,
                        float* __restrict__ outp)
{
    const int n  = blockIdx.x * 256 + threadIdx.x;   // 0 .. NPIX-1
    const int g  = blockIdx.y;                       // head == group
    const int b  = n / HWP;
    const int hw = n - b * HWP;
    const int h  = hw / WW;
    const int w  = hw - h * WW;

    const size_t gbase = ((size_t)b * DG + g) * HWP * CG;
    const float* kb = kp + gbase;
    const float* vb = vp + gbase;

    // q vector (contiguous 64B)
    float4 qv[4];
    {
        const float4* q4 = (const float4*)(qp + gbase + (size_t)hw * CG);
        #pragma unroll
        for (int j = 0; j < 4; ++j) qv[j] = q4[j];
    }

    // sample coordinates
    float pys[KK], pxs[KK], sc[KK];
    const size_t obase = ((size_t)b * (DG * 2 * KK)) * HWP + hw;
    #pragma unroll
    for (int k = 0; k < KK; ++k) {
        int ki = k / KW, kj = k - ki * KW;
        float oy = offs[obase + (size_t)((g * KK + k) * 2 + 0) * HWP];
        float ox = offs[obase + (size_t)((g * KK + k) * 2 + 1) * HWP];
        pys[k] = (float)(h - PADR + ki) + oy;
        pxs[k] = (float)(w - PADR + kj) + ox;
    }

    // ---- pass 1: scores ----
    #pragma unroll
    for (int k = 0; k < KK; ++k) {
        float py = pys[k], px = pxs[k];
        float y0 = floorf(py), x0 = floorf(px);
        float wy1 = py - y0, wx1 = px - x0;
        float wy0 = 1.0f - wy1, wx0 = 1.0f - wx1;
        float s = 0.0f;
        #pragma unroll
        for (int c = 0; c < 4; ++c) {
            float yy = y0 + (float)(c >> 1);
            float xx = x0 + (float)(c & 1);
            float wt = ((c >> 1) ? wy1 : wy0) * ((c & 1) ? wx1 : wx0);
            bool vld = (yy >= 0.0f) && (yy <= (float)(HH - 1)) &&
                       (xx >= 0.0f) && (xx <= (float)(WW - 1));
            float we = vld ? wt : 0.0f;
            if (we != 0.0f) {
                int yc = (int)fminf(fmaxf(yy, 0.0f), (float)(HH - 1));
                int xc = (int)fminf(fmaxf(xx, 0.0f), (float)(WW - 1));
                const float4* k4 = (const float4*)(kb + (size_t)(yc * WW + xc) * CG);
                float dot = 0.0f;
                #pragma unroll
                for (int j = 0; j < 4; ++j) {
                    float4 kvv = k4[j];
                    dot += qv[j].x * kvv.x + qv[j].y * kvv.y +
                           qv[j].z * kvv.z + qv[j].w * kvv.w;
                }
                s += we * dot;
            }
        }
        sc[k] = s * 0.25f;   // hd^-0.5, hd = 16
    }

    // ---- softmax over K ----
    float mx = sc[0];
    #pragma unroll
    for (int k = 1; k < KK; ++k) mx = fmaxf(mx, sc[k]);
    float ssum = 0.0f;
    #pragma unroll
    for (int k = 0; k < KK; ++k) { sc[k] = expf(sc[k] - mx); ssum += sc[k]; }
    float rs = 1.0f / ssum;
    #pragma unroll
    for (int k = 0; k < KK; ++k) sc[k] *= rs;

    // ---- pass 2: attn-weighted v gather ----
    float4 ov[4];
    #pragma unroll
    for (int j = 0; j < 4; ++j) ov[j] = make_float4(0.f, 0.f, 0.f, 0.f);

    #pragma unroll
    for (int k = 0; k < KK; ++k) {
        float py = pys[k], px = pxs[k];
        float y0 = floorf(py), x0 = floorf(px);
        float wy1 = py - y0, wx1 = px - x0;
        float wy0 = 1.0f - wy1, wx0 = 1.0f - wx1;
        #pragma unroll
        for (int c = 0; c < 4; ++c) {
            float yy = y0 + (float)(c >> 1);
            float xx = x0 + (float)(c & 1);
            float wt = ((c >> 1) ? wy1 : wy0) * ((c & 1) ? wx1 : wx0);
            bool vld = (yy >= 0.0f) && (yy <= (float)(HH - 1)) &&
                       (xx >= 0.0f) && (xx <= (float)(WW - 1));
            float we = vld ? (wt * sc[k]) : 0.0f;
            if (we != 0.0f) {
                int yc = (int)fminf(fmaxf(yy, 0.0f), (float)(HH - 1));
                int xc = (int)fminf(fmaxf(xx, 0.0f), (float)(WW - 1));
                const float4* v4 = (const float4*)(vb + (size_t)(yc * WW + xc) * CG);
                #pragma unroll
                for (int j = 0; j < 4; ++j) {
                    float4 vv = v4[j];
                    ov[j].x += we * vv.x; ov[j].y += we * vv.y;
                    ov[j].z += we * vv.z; ov[j].w += we * vv.w;
                }
            }
        }
    }

    // output stays channel-major [B, C, HW] for the MLP GEMM loader
    const size_t cbase = ((size_t)b * CC + g * CG) * HWP + hw;
    #pragma unroll
    for (int j = 0; j < 4; ++j) {
        outp[cbase + (size_t)(4*j + 0) * HWP] = ov[j].x;
        outp[cbase + (size_t)(4*j + 1) * HWP] = ov[j].y;
        outp[cbase + (size_t)(4*j + 2) * HWP] = ov[j].z;
        outp[cbase + (size_t)(4*j + 3) * HWP] = ov[j].w;
    }
}

// ---------------------------------------------------------------------------
// Launch: q/k/v projections (WMMA, group-pixel-major out) -> deform-attn
// (float4 gathers) -> MLP GEMM1+GELU (WMMA) -> MLP GEMM2 + residual -> d_out
// ---------------------------------------------------------------------------
extern "C" void kernel_launch(void* const* d_in, const int* in_sizes, int n_in,
                              void* d_out, int out_size, void* d_ws, size_t ws_size,
                              hipStream_t stream) {
    const float* q    = (const float*)d_in[0];
    const float* k    = (const float*)d_in[1];
    const float* v    = (const float*)d_in[2];
    const float* offs = (const float*)d_in[3];
    const float* Wq   = (const float*)d_in[4];
    const float* bq   = (const float*)d_in[5];
    const float* Wk   = (const float*)d_in[6];
    const float* bk   = (const float*)d_in[7];
    const float* Wv   = (const float*)d_in[8];
    const float* bv   = (const float*)d_in[9];
    const float* W1   = (const float*)d_in[10];
    const float* b1   = (const float*)d_in[11];
    const float* W2   = (const float*)d_in[12];
    const float* b2   = (const float*)d_in[13];
    float* out = (float*)d_out;

    // workspace carve-up (~85 MB)
    float* qp   = (float*)d_ws;
    float* kp   = qp   + (size_t)NPIX * CC;
    float* vp   = kp   + (size_t)NPIX * CC;
    float* attn = vp   + (size_t)NPIX * CC;
    float* h1   = attn + (size_t)NPIX * CC;       // [B, 384, HW]

    dim3 blk(256);
    dim3 g192(NPIX / 128, 1);                     // 144 M-tiles, Cout=192
    dim3 g384(NPIX / 128, 2);                     // Cout=384 -> two N-blocks

    // 1-3: q/k/v projections, output in [B, G, HW, 16] for fast gathers
    gemm_nchw_wmma<<<g192, blk, 0, stream>>>(q, CC, Wq, bq, CC, qp, nullptr, 0, 1);
    gemm_nchw_wmma<<<g192, blk, 0, stream>>>(k, CC, Wk, bk, CC, kp, nullptr, 0, 1);
    gemm_nchw_wmma<<<g192, blk, 0, stream>>>(v, CC, Wv, bv, CC, vp, nullptr, 0, 1);

    // 4: deformable sampling + attention (channel-major output)
    dim3 ga(NPIX / 256, HEADS);                   // (72, 12)
    deform_attn_kernel<<<ga, blk, 0, stream>>>(qp, kp, vp, offs, attn);

    // 5: MLP hidden (GELU exact)
    gemm_nchw_wmma<<<g384, blk, 0, stream>>>(attn, CC, W1, b1, MLP_H, h1, nullptr, 1, 0);

    // 6: MLP out + residual -> final output
    gemm_nchw_wmma<<<g192, blk, 0, stream>>>(h1, MLP_H, W2, b2, CC, out, attn, 0, 0);
}
